// ContrastiveRPN_64518998720642
// MI455X (gfx1250) — compile-verified, matrix operand order checked by reading, not run on hardware
//
#include <hip/hip_runtime.h>
#include <hip/hip_bf16.h>
#include <math.h>

// ---------------- constants (match reference) ----------------
#define B_N    32
#define CLIP   1024
#define NPOS   256          // 16x16 spatial positions
#define OCH    512
#define NA     9
#define TOPK   30
#define NANCH  (NPOS*NA)    // 2304
#define NROI   (B_N*TOPK)   // 960
#define PROJD  256
#define KCONV  (9*CLIP)     // 9216
#define KT_CONV (KCONV/32)  // 288
#define NMS_TH 0.85f
#define TEMPER 0.07f
#define LDS_RS 1032         // padded row stride (halves): 1024 + 8 -> bank-staggered

typedef __attribute__((ext_vector_type(16))) __bf16 v16bf;
typedef __attribute__((ext_vector_type(8)))  __bf16 v8bf;
typedef __attribute__((ext_vector_type(8)))  float  v8f;

// ---------------- bf16 helpers (bit-exact RNE) ----------------
__device__ __forceinline__ __bf16 f2bf(float f) {
    unsigned u = __float_as_uint(f);
    u = (u + 0x7FFFu + ((u >> 16) & 1u)) >> 16;
    unsigned short s = (unsigned short)u;
    return __builtin_bit_cast(__bf16, s);
}
__device__ __forceinline__ float bf2f(__bf16 b) {
    unsigned short s = __builtin_bit_cast(unsigned short, b);
    return __uint_as_float(((unsigned)s) << 16);
}

// K-index within a 32-wide K slice for 16-bit A/B fragments (ISA 7.12.2):
// lanes 0-15 hold K {0..7,16..23}, lanes 16-31 hold K {8..15,24..31}
__device__ __forceinline__ int frag_k(int lane, int e) {
    int half = (lane >> 4) & 1;
    return (e < 8) ? (half * 8 + e) : (8 + half * 8 + e);
}

// Load a 16x32 bf16 A-fragment from a row-major tile (tile = &A[row0*lda + k0])
__device__ __forceinline__ v16bf load_A_frag(const __bf16* tile, int lda) {
    int lane = threadIdx.x & 31;
    const __bf16* row = tile + (size_t)(lane & 15) * lda + ((lane >> 4) * 8);
    v8bf lo = *(const v8bf*)row;
    v8bf hi = *(const v8bf*)(row + 16);
    v16bf r;
#pragma unroll
    for (int e = 0; e < 8; ++e) { r[e] = lo[e]; r[e + 8] = hi[e]; }
    return r;
}

// Load a pre-swizzled 32x16 B-fragment (512 halves, fragment-linear layout)
__device__ __forceinline__ v16bf load_B_frag(const __bf16* frag) {
    return *(const v16bf*)(frag + (size_t)(threadIdx.x & 31) * 16);
}

// ======================= packing kernels =======================

// clip_features (257, B, 1024) f32  ->  sp[b][p][c] bf16 (drop CLS token)
__global__ void k_pack_spatial(const float* __restrict__ clip, __bf16* __restrict__ sp) {
    int tid = blockIdx.x * blockDim.x + threadIdx.x;
    if (tid >= B_N * NPOS * CLIP) return;
    int c = tid & (CLIP - 1);
    int rest = tid >> 10;
    int p = rest & 255;
    int b = rest >> 8;
    sp[tid] = f2bf(clip[((size_t)(1 + p) * B_N + b) * CLIP + c]);
}

// conv_w (512,1024,3,3) OIHW f32 -> fragment-linear bf16 (K order: tap*1024+c)
__global__ void k_pack_convw(const float* __restrict__ w, __bf16* __restrict__ wB) {
    int tid = blockIdx.x * blockDim.x + threadIdx.x;
    if (tid >= (OCH / 16) * KT_CONV * 512) return;
    int e = tid & 15, lane = (tid >> 4) & 31, f = tid >> 9;
    int kt = f % KT_CONV, nt = f / KT_CONV;
    int k = kt * 32 + frag_k(lane, e);
    int n = nt * 16 + (lane & 15);
    int tap = k >> 10, c = k & 1023;
    wB[tid] = f2bf(w[((size_t)n * CLIP + c) * 9 + tap]);
}

// cls_w (9,512) + reg_w (36,512) -> combined padded B (K=512, N=48)
__global__ void k_pack_wcr(const float* __restrict__ cls_w, const float* __restrict__ reg_w,
                           __bf16* __restrict__ wB) {
    int tid = blockIdx.x * blockDim.x + threadIdx.x;
    if (tid >= 3 * 16 * 512) return;
    int e = tid & 15, lane = (tid >> 4) & 31, f = tid >> 9;
    int kt = f % 16, nt = f / 16;
    int k = kt * 32 + frag_k(lane, e);
    int n = nt * 16 + (lane & 15);
    float v = 0.f;
    if (n < NA)       v = cls_w[(size_t)n * OCH + k];
    else if (n < 45)  v = reg_w[(size_t)(n - NA) * OCH + k];
    wB[tid] = f2bf(v);
}

// proj_w1 (256,1024) -> B frags (K=1024, N=256)
__global__ void k_pack_w1(const float* __restrict__ w, __bf16* __restrict__ wB) {
    int tid = blockIdx.x * blockDim.x + threadIdx.x;
    if (tid >= 16 * 32 * 512) return;
    int e = tid & 15, lane = (tid >> 4) & 31, f = tid >> 9;
    int kt = f % 32, nt = f / 32;
    int k = kt * 32 + frag_k(lane, e);
    int n = nt * 16 + (lane & 15);
    wB[tid] = f2bf(w[(size_t)n * 1024 + k]);
}

// proj_w2 (256,256) -> B frags (K=256, N=256)
__global__ void k_pack_w2(const float* __restrict__ w, __bf16* __restrict__ wB) {
    int tid = blockIdx.x * blockDim.x + threadIdx.x;
    if (tid >= 16 * 8 * 512) return;
    int e = tid & 15, lane = (tid >> 4) & 31, f = tid >> 9;
    int kt = f % 8, nt = f / 8;
    int k = kt * 32 + frag_k(lane, e);
    int n = nt * 16 + (lane & 15);
    wB[tid] = f2bf(w[(size_t)n * 256 + k]);
}

// ======================= conv (implicit GEMM, WMMA + async-LDS) ===========
// grid (16 posRows, 1, 32 batches), block 256 = 8 waves.
// Each wave owns 4 oc-tiles (4 independent accumulators -> back-to-back WMMAs).
// Activation rows staged in LDS via global_load_async_to_lds_b128 (ASYNCcnt),
// with zero-filled halo columns so the inner loop is branchless.
__global__ void k_conv_wmma(const __bf16* __restrict__ sp, const __bf16* __restrict__ wB,
                            const float* __restrict__ conv_b, __bf16* __restrict__ xout) {
    __shared__ __bf16 abuf[18][LDS_RS];    // [pj+1][channel], rows 0 & 17 = zero halo
    int b = blockIdx.z;
    int pt = blockIdx.x;                   // feature-map row i
    int wave = threadIdx.x >> 5;
    int lane = threadIdx.x & 31;
    int ot0 = wave * 4;                    // first of 4 oc-tiles for this wave
    v8f acc[4] = {};

    for (int dyy = 0; dyy < 3; ++dyy) {
        int pi = pt - 1 + dyy;
        bool rowvalid = (pi >= 0) && (pi < 16);   // uniform across block

        // ---- zero the staging buffer (halo + padding rows) ----
        {
            int4* z = (int4*)&abuf[0][0];
            const int tot = (18 * LDS_RS * 2) / 16;     // 2322 16B chunks
            for (int i = threadIdx.x; i < tot; i += 256) z[i] = make_int4(0, 0, 0, 0);
        }
        __syncthreads();

        // ---- async copy 16 positions x 1024 ch into interior rows 1..16 ----
        if (rowvalid) {
#pragma unroll
            for (int q = 0; q < 8; ++q) {
                int ch = threadIdx.x + q * 256;         // 0..2047 16B chunks
                int j = ch >> 7, c16 = ch & 127;
                unsigned ldsb = (unsigned)(unsigned long long)(void*)&abuf[j + 1][c16 * 8];
                unsigned long long ga = (unsigned long long)(const void*)
                    (sp + ((size_t)(b * NPOS + pi * 16 + j) * CLIP + c16 * 8));
                asm volatile("global_load_async_to_lds_b128 %0, %1, off"
                             :: "v"(ldsb), "v"(ga) : "memory");
            }
            asm volatile("s_wait_asynccnt 0x0" ::: "memory");
        }
        __syncthreads();

        // ---- 3 horizontal taps against this row-block ----
        for (int dx = -1; dx <= 1; ++dx) {
            int tap = dyy * 3 + (dx + 1);
            const __bf16* rowp = &abuf[(lane & 15) + dx + 1][0] + ((lane >> 4) * 8);
#pragma unroll 1
            for (int kc = 0; kc < CLIP; kc += 32) {
                v8bf lo = *(const v8bf*)(rowp + kc);
                v8bf hi = *(const v8bf*)(rowp + kc + 16);
                v16bf a;
#pragma unroll
                for (int e = 0; e < 8; ++e) { a[e] = lo[e]; a[e + 8] = hi[e]; }
                int kt = tap * 32 + (kc >> 5);
                const __bf16* fb = wB + ((size_t)(ot0 * KT_CONV + kt) * 512);
                if (kt + 1 < KT_CONV)
                    __builtin_prefetch(fb + 512 + (size_t)lane * 16, 0, 3);
#pragma unroll
                for (int tt = 0; tt < 4; ++tt) {
                    v16bf bb = load_B_frag(fb + (size_t)tt * KT_CONV * 512);
                    acc[tt] = __builtin_amdgcn_wmma_f32_16x16x32_bf16(
                        false, a, false, bb, (short)0, acc[tt], false, false);
                }
            }
        }
        __syncthreads();   // before re-zeroing for the next row-block
    }

    // ---- epilogue: bias + ReLU, store 4 oc-tiles ----
#pragma unroll
    for (int tt = 0; tt < 4; ++tt) {
        int oc = (ot0 + tt) * 16 + (lane & 15);
        float bias = conv_b[oc];
#pragma unroll
        for (int r = 0; r < 8; ++r) {
            int m = r + 8 * (lane >> 4);               // position j within row pt
            float v = acc[tt][r] + bias;
            v = v > 0.f ? v : 0.f;                      // ReLU
            xout[(size_t)(b * NPOS + pt * 16 + m) * OCH + oc] = f2bf(v);
        }
    }
}

// ======================= cls/reg head (WMMA) =======================
// grid (16 posRows, 3 nTiles, 32 batches), block 32
__global__ void k_clsreg(const __bf16* __restrict__ x, const __bf16* __restrict__ wB,
                         const float* __restrict__ cls_b, const float* __restrict__ reg_b,
                         float* __restrict__ scores, float* __restrict__ regp) {
    int b = blockIdx.z, pt = blockIdx.x, nt = blockIdx.y;
    int lane = threadIdx.x & 31;
    v8f acc = {};
    const __bf16* Abase = x + (size_t)(b * NPOS + pt * 16) * OCH;
#pragma unroll 1
    for (int kt = 0; kt < 16; ++kt) {
        v16bf a = load_A_frag(Abase + kt * 32, OCH);
        v16bf bb = load_B_frag(wB + (size_t)(nt * 16 + kt) * 512);
        acc = __builtin_amdgcn_wmma_f32_16x16x32_bf16(
            false, a, false, bb, (short)0, acc, false, false);
    }
    int n = lane & 15, gn = nt * 16 + n;
#pragma unroll
    for (int r = 0; r < 8; ++r) {
        int m = r + 8 * (lane >> 4);
        int p = pt * 16 + m;
        float v = acc[r];
        if (gn < NA) {
            v += cls_b[gn];
            scores[(size_t)b * NANCH + p * NA + gn] = 1.f / (1.f + expf(-v));
        } else if (gn < 45) {
            int q = gn - NA, aa = q >> 2, comp = q & 3;
            v += reg_b[q];
            regp[((size_t)b * NANCH + p * NA + aa) * 4 + comp] = v;
        }
    }
}

// ======================= anchor decode =======================
__global__ void k_decode(const float* __restrict__ regp, float* __restrict__ boxes) {
    int tid = blockIdx.x * blockDim.x + threadIdx.x;
    if (tid >= B_N * NANCH) return;
    int idx = tid % NANCH;
    int p = idx / NA, a = idx % NA;
    int i = p >> 4, j = p & 15;
    const float scales[3] = {8.f, 16.f, 32.f};
    const float ratios[3] = {0.5f, 1.f, 2.f};
    float s = scales[a / 3], r = ratios[a % 3];
    float aw = s * sqrtf(r), ah = s / sqrtf(r);
    float xc = (j + 0.5f) * 16.f, yc = (i + 0.5f) * 16.f;
    const float* rg = regp + (size_t)tid * 4;
    float cx = rg[0] * aw + xc, cy = rg[1] * ah + yc;
    float pw = expf(rg[2]) * aw, ph = expf(rg[3]) * ah;
    float x1 = fminf(fmaxf(cx - 0.5f * pw, 0.f), 256.f);
    float y1 = fminf(fmaxf(cy - 0.5f * ph, 0.f), 256.f);
    float x2 = fminf(fmaxf(cx + 0.5f * pw, 0.f), 256.f);
    float y2 = fminf(fmaxf(cy + 0.5f * ph, 0.f), 256.f);
    if (x2 - x1 < 1.f) x2 = x1 + 1.f;
    if (y2 - y1 < 1.f) y2 = y1 + 1.f;
    float* bo = boxes + (size_t)tid * 4;
    bo[0] = x1; bo[1] = y1; bo[2] = x2; bo[3] = y2;
}

// ======================= top-K (iterative argmax, LDS) =======================
__global__ void k_topk(const float* __restrict__ scores, const float* __restrict__ boxes,
                       float* __restrict__ rois) {
    __shared__ float sv[NANCH];
    __shared__ float rv[256];
    __shared__ int   ri[256];
    int b = blockIdx.x, t = threadIdx.x;
#pragma unroll
    for (int q = 0; q < 9; ++q) sv[t + q * 256] = scores[(size_t)b * NANCH + t + q * 256];
    __syncthreads();
    for (int k = 0; k < TOPK; ++k) {
        float bv = -3.4e38f; int bi = 0;
#pragma unroll
        for (int q = 0; q < 9; ++q) {
            int idx = t + q * 256;
            float v = sv[idx];
            if (v > bv) { bv = v; bi = idx; }   // ascending idx -> lowest idx on ties
        }
        rv[t] = bv; ri[t] = bi; __syncthreads();
        for (int s = 128; s > 0; s >>= 1) {
            if (t < s) {
                if (rv[t + s] > rv[t] || (rv[t + s] == rv[t] && ri[t + s] < ri[t])) {
                    rv[t] = rv[t + s]; ri[t] = ri[t + s];
                }
            }
            __syncthreads();
        }
        if (t == 0) {
            int best = ri[0];
            sv[best] = -3.4e38f;
            for (int c = 0; c < 4; ++c)
                rois[((size_t)b * TOPK + k) * 4 + c] = boxes[((size_t)b * NANCH + best) * 4 + c];
        }
        __syncthreads();
    }
}

// ======================= NMS: one wave32 per batch, shuffle broadcast ======
__global__ void k_nms(const float* __restrict__ rois, int* __restrict__ valid) {
    int b = blockIdx.x, k = threadIdx.x;   // 32 lanes, K=30 boxes
    float x1 = 0.f, y1 = 0.f, x2 = 1.f, y2 = 1.f;
    if (k < TOPK) {
        const float* r = rois + ((size_t)b * TOPK + k) * 4;
        x1 = r[0]; y1 = r[1]; x2 = r[2]; y2 = r[3];
    }
    float area = (x2 - x1) * (y2 - y1);
    int keep = (k < TOPK) ? 1 : 0;
    for (int i = 0; i < TOPK; ++i) {
        int   ki  = __shfl(keep, i);
        float bx1 = __shfl(x1, i), by1 = __shfl(y1, i);
        float bx2 = __shfl(x2, i), by2 = __shfl(y2, i);
        float ai  = __shfl(area, i);
        float iw = fmaxf(fminf(x2, bx2) - fmaxf(x1, bx1), 0.f);
        float ih = fmaxf(fminf(y2, by2) - fmaxf(y1, by1), 0.f);
        float inter = iw * ih;
        float iou = inter / (area + ai - inter);
        if (k > i && ki && iou > NMS_TH) keep = 0;
    }
    if (k < TOPK) valid[b * TOPK + k] = keep;
}

// ======================= ROI-align mean pool (7x7 bilinear) ================
__global__ void k_roialign(const __bf16* __restrict__ sp, const float* __restrict__ rois,
                           __bf16* __restrict__ pooled) {
    int n = blockIdx.x, t = threadIdx.x;
    int b = n / TOPK;
    const float* r = rois + (size_t)n * 4;
    float rx1 = r[0] * (1.f / 16.f), ry1 = r[1] * (1.f / 16.f);
    float rw = fmaxf(r[2] * (1.f / 16.f) - rx1, 1.f);
    float rh = fmaxf(r[3] * (1.f / 16.f) - ry1, 1.f);
    float acc[4] = {0.f, 0.f, 0.f, 0.f};
    for (int s = 0; s < 49; ++s) {
        int py = s / 7, px = s % 7;
        float X = rx1 + (px + 0.5f) * (rw / 7.f);
        float Y = ry1 + (py + 0.5f) * (rh / 7.f);
        bool ok = (Y > -1.f) && (Y < 16.f) && (X > -1.f) && (X < 16.f);
        if (!ok) continue;
        float Xc = fminf(fmaxf(X, 0.f), 15.f);
        float Yc = fminf(fmaxf(Y, 0.f), 15.f);
        int x0 = (int)floorf(Xc), y0 = (int)floorf(Yc);
        int x1i = min(x0 + 1, 15), y1i = min(y0 + 1, 15);
        float lx = Xc - (float)x0, ly = Yc - (float)y0;
        float hx = 1.f - lx, hy = 1.f - ly;
        float w00 = hy * hx, w01 = hy * lx, w10 = ly * hx, w11 = ly * lx;
        const __bf16* f00 = sp + (size_t)(b * NPOS + y0 * 16 + x0) * CLIP;
        const __bf16* f01 = sp + (size_t)(b * NPOS + y0 * 16 + x1i) * CLIP;
        const __bf16* f10 = sp + (size_t)(b * NPOS + y1i * 16 + x0) * CLIP;
        const __bf16* f11 = sp + (size_t)(b * NPOS + y1i * 16 + x1i) * CLIP;
#pragma unroll
        for (int q = 0; q < 4; ++q) {
            int c = t + q * 256;
            acc[q] += w00 * bf2f(f00[c]) + w01 * bf2f(f01[c]) +
                      w10 * bf2f(f10[c]) + w11 * bf2f(f11[c]);
        }
    }
#pragma unroll
    for (int q = 0; q < 4; ++q)
        pooled[(size_t)n * CLIP + t + q * 256] = f2bf(acc[q] * (1.f / 49.f));
}

// ======================= integral image (padded 257x257) ===================
__global__ void k_integral(const float* __restrict__ gt, float* __restrict__ Ipad) {
    int b = blockIdx.x, t = threadIdx.x;   // 256 threads
    float* I = Ipad + (size_t)b * 257 * 257;
    I[t] = 0.f;                            // row 0
    if (t == 0) I[256] = 0.f;
    I[(size_t)(t + 1) * 257] = 0.f;        // col 0
    {   // row-wise cumsum (thread t = image row)
        const float* g = gt + (size_t)b * 256 * 256 + (size_t)t * 256;
        float* row = I + (size_t)(t + 1) * 257 + 1;
        float run = 0.f;
        for (int x = 0; x < 256; ++x) { run += g[x]; row[x] = run; }
    }
    __syncthreads();
    {   // column-wise cumsum (thread t = image col)
        float run = 0.f;
        for (int y = 1; y <= 256; ++y) {
            float* cell = I + (size_t)y * 257 + (t + 1);
            run += *cell; *cell = run;
        }
    }
}

__global__ void k_labels(const float* __restrict__ rois, const float* __restrict__ Ipad,
                         int* __restrict__ labels) {
    int n = blockIdx.x * blockDim.x + threadIdx.x;
    if (n >= NROI) return;
    int b = n / TOPK;
    const float* r = rois + (size_t)n * 4;
    int x1 = min(max((int)r[0], 0), 256);
    int y1 = min(max((int)r[1], 0), 256);
    int x2 = min(max((int)r[2], 0), 256);
    int y2 = min(max((int)r[3], 0), 256);
    const float* I = Ipad + (size_t)b * 257 * 257;
    float s = I[(size_t)y2 * 257 + x2] - I[(size_t)y1 * 257 + x2]
            - I[(size_t)y2 * 257 + x1] + I[(size_t)y1 * 257 + x1];
    int cnt = (y2 - y1) * (x2 - x1);
    labels[n] = (cnt > 0 && (s / (float)max(cnt, 1) > 0.5f)) ? 1 : 0;
}

// ======================= generic WMMA GEMM (MLP1/MLP2/sim) =================
// grid (mt, nt), block 32. A row-major bf16 (lda=K), wB fragment-linear.
__global__ void k_gemm(const __bf16* __restrict__ A, const __bf16* __restrict__ wB,
                       const float* __restrict__ bias, int lda, int KT, int N,
                       float scale, int relu,
                       __bf16* __restrict__ out_bf, float* __restrict__ out_f) {
    int mt = blockIdx.x, nt = blockIdx.y;
    int lane = threadIdx.x & 31;
    v8f acc = {};
    const __bf16* Abase = A + (size_t)mt * 16 * lda;
#pragma unroll 1
    for (int kt = 0; kt < KT; ++kt) {
        v16bf a  = load_A_frag(Abase + kt * 32, lda);
        v16bf bb = load_B_frag(wB + (size_t)(nt * KT + kt) * 512);
        acc = __builtin_amdgcn_wmma_f32_16x16x32_bf16(
            false, a, false, bb, (short)0, acc, false, false);
    }
    int n = lane & 15, gn = nt * 16 + n;
    float bv = bias ? bias[gn] : 0.f;
#pragma unroll
    for (int r = 0; r < 8; ++r) {
        int m = mt * 16 + r + 8 * (lane >> 4);
        float v = acc[r] * scale + bv;
        if (relu) v = fmaxf(v, 0.f);
        if (out_bf) out_bf[(size_t)m * N + gn] = f2bf(v);
        if (out_f)  out_f[(size_t)m * N + gn]  = v;
    }
}

// ======================= row L2 norms =======================
__global__ void k_rownorm(const float* __restrict__ proj, float* __restrict__ invn) {
    __shared__ float red[256];
    int n = blockIdx.x, t = threadIdx.x;
    float v = proj[(size_t)n * PROJD + t];
    red[t] = v * v; __syncthreads();
    for (int s = 128; s > 0; s >>= 1) { if (t < s) red[t] += red[t + s]; __syncthreads(); }
    if (t == 0) invn[n] = 1.f / fmaxf(sqrtf(red[0]), 1e-12f);
}

// normalized features: row-major A and fragment-linear B (for nf @ nf.T)
__global__ void k_nfA(const float* __restrict__ proj, const float* __restrict__ invn,
                      __bf16* __restrict__ nfA) {
    int tid = blockIdx.x * blockDim.x + threadIdx.x;
    if (tid >= NROI * PROJD) return;
    int n = tid >> 8;
    nfA[tid] = f2bf(proj[tid] * invn[n]);
}
__global__ void k_nfB(const float* __restrict__ proj, const float* __restrict__ invn,
                      __bf16* __restrict__ nfB) {
    int tid = blockIdx.x * blockDim.x + threadIdx.x;
    if (tid >= 60 * 8 * 512) return;
    int e = tid & 15, lane = (tid >> 4) & 31, f = tid >> 9;
    int kt = f % 8, nt = f / 8;
    int k = kt * 32 + frag_k(lane, e);
    int n = nt * 16 + (lane & 15);
    nfB[tid] = f2bf(proj[(size_t)n * PROJD + k] * invn[n]);
}

// ======================= SupCon loss =======================
__global__ void k_lossrow(const float* __restrict__ sim, const int* __restrict__ labels,
                          const int* __restrict__ valid,
                          float* __restrict__ lrow, float* __restrict__ arow) {
    __shared__ float rn[256], rd[256];
    __shared__ int rp[256];
    int i = blockIdx.x, t = threadIdx.x;
    int li = labels[i], vi = valid[i];
    float num = 0.f, den = 0.f; int pc = 0;
    for (int q = 0; q < 4; ++q) {
        int j = t + q * 256;
        if (j >= NROI) break;
        float es = expf(sim[(size_t)i * NROI + j]);
        bool nd  = (j != i);
        bool vv  = vi && valid[j];
        bool msk = nd && vv;
        bool pos = msk && (labels[j] == li);
        den += msk ? es : 0.f;
        num += pos ? es : 0.f;
        pc  += pos ? 1 : 0;
    }
    rn[t] = num; rd[t] = den; rp[t] = pc; __syncthreads();
    for (int s = 128; s > 0; s >>= 1) {
        if (t < s) { rn[t] += rn[t + s]; rd[t] += rd[t + s]; rp[t] += rp[t + s]; }
        __syncthreads();
    }
    if (t == 0) {
        float ratio = rn[0] / (rd[0] + 1e-12f);
        float l = -logf(ratio + 1e-12f);
        int active = vi && (rp[0] > 0);
        lrow[i] = active ? l : 0.f;
        arow[i] = active ? 1.f : 0.f;
    }
}

__global__ void k_final(const float* __restrict__ lrow, const float* __restrict__ arow,
                        float* __restrict__ out) {
    __shared__ float rl[256], ra[256];
    int t = threadIdx.x;
    float sl = 0.f, sa = 0.f;
    for (int q = 0; q < 4; ++q) {
        int i = t + q * 256;
        if (i < NROI) { sl += lrow[i]; sa += arow[i]; }
    }
    rl[t] = sl; ra[t] = sa; __syncthreads();
    for (int s = 128; s > 0; s >>= 1) {
        if (t < s) { rl[t] += rl[t + s]; ra[t] += ra[t + s]; }
        __syncthreads();
    }
    if (t == 0) out[0] = (ra[0] > 0.f) ? (rl[0] / fmaxf(ra[0], 1.f)) : 0.f;
}

// ======================= host launch =======================
extern "C" void kernel_launch(void* const* d_in, const int* in_sizes, int n_in,
                              void* d_out, int out_size, void* d_ws, size_t ws_size,
                              hipStream_t stream) {
    const float* clip   = (const float*)d_in[0];
    const float* gt     = (const float*)d_in[1];
    const float* conv_w = (const float*)d_in[2];
    const float* conv_b = (const float*)d_in[3];
    const float* cls_w  = (const float*)d_in[4];
    const float* cls_b  = (const float*)d_in[5];
    const float* reg_w  = (const float*)d_in[6];
    const float* reg_b  = (const float*)d_in[7];
    const float* w1     = (const float*)d_in[8];
    const float* b1     = (const float*)d_in[9];
    const float* w2     = (const float*)d_in[10];
    const float* b2     = (const float*)d_in[11];
    float* out = (float*)d_out;

    char* p = (char*)d_ws;
    auto carve = [&](size_t bytes) -> char* {
        char* r = p; p += (bytes + 255) & ~(size_t)255; return r;
    };
    __bf16* sp     = (__bf16*)carve((size_t)B_N * NPOS * CLIP * 2);
    __bf16* wconvB = (__bf16*)carve((size_t)KCONV * OCH * 2);
    __bf16* xbf    = (__bf16*)carve((size_t)B_N * NPOS * OCH * 2);
    __bf16* wcrB   = (__bf16*)carve((size_t)512 * 48 * 2);
    float*  scores = (float*)carve((size_t)B_N * NANCH * 4);
    float*  regp   = (float*)carve((size_t)B_N * NANCH * 4 * 4);
    float*  boxes  = (float*)carve((size_t)B_N * NANCH * 4 * 4);
    float*  rois   = (float*)carve((size_t)NROI * 4 * 4);
    int*    valid  = (int*)carve((size_t)NROI * 4);
    __bf16* pooled = (__bf16*)carve((size_t)NROI * CLIP * 2);
    __bf16* w1B    = (__bf16*)carve((size_t)1024 * 256 * 2);
    __bf16* w2B    = (__bf16*)carve((size_t)256 * 256 * 2);
    __bf16* hbf    = (__bf16*)carve((size_t)NROI * PROJD * 2);
    float*  proj   = (float*)carve((size_t)NROI * PROJD * 4);
    float*  invn   = (float*)carve((size_t)NROI * 4);
    __bf16* nfA    = (__bf16*)carve((size_t)NROI * PROJD * 2);
    __bf16* nfB    = (__bf16*)carve((size_t)60 * 8 * 512 * 2);
    float*  sim    = (float*)carve((size_t)NROI * NROI * 4);
    float*  Ipad   = (float*)carve((size_t)B_N * 257 * 257 * 4);
    int*    labels = (int*)carve((size_t)NROI * 4);
    float*  lrow   = (float*)carve((size_t)NROI * 4);
    float*  arow   = (float*)carve((size_t)NROI * 4);

    // --- packing ---
    k_pack_spatial<<<(B_N * NPOS * CLIP + 255) / 256, 256, 0, stream>>>(clip, sp);
    k_pack_convw<<<((OCH / 16) * KT_CONV * 512 + 255) / 256, 256, 0, stream>>>(conv_w, wconvB);
    k_pack_wcr<<<(3 * 16 * 512 + 255) / 256, 256, 0, stream>>>(cls_w, reg_w, wcrB);
    k_pack_w1<<<(16 * 32 * 512 + 255) / 256, 256, 0, stream>>>(w1, w1B);
    k_pack_w2<<<(16 * 8 * 512 + 255) / 256, 256, 0, stream>>>(w2, w2B);

    // --- conv + heads ---
    k_conv_wmma<<<dim3(16, 1, B_N), 256, 0, stream>>>(sp, wconvB, conv_b, xbf);
    k_clsreg<<<dim3(16, 3, B_N), 32, 0, stream>>>(xbf, wcrB, cls_b, reg_b, scores, regp);
    k_decode<<<(B_N * NANCH + 255) / 256, 256, 0, stream>>>(regp, boxes);

    // --- topk / nms / pooling / labels ---
    k_topk<<<B_N, 256, 0, stream>>>(scores, boxes, rois);
    k_nms<<<B_N, 32, 0, stream>>>(rois, valid);
    k_roialign<<<NROI, 256, 0, stream>>>(sp, rois, pooled);
    k_integral<<<B_N, 256, 0, stream>>>(gt, Ipad);
    k_labels<<<(NROI + 255) / 256, 256, 0, stream>>>(rois, Ipad, labels);

    // --- projection MLP (WMMA) ---
    k_gemm<<<dim3(60, 16), 32, 0, stream>>>(pooled, w1B, b1, 1024, 32, 256,
                                            1.f, 1, hbf, nullptr);
    k_gemm<<<dim3(60, 16), 32, 0, stream>>>(hbf, w2B, b2, 256, 8, 256,
                                            1.f, 0, nullptr, proj);

    // --- SupCon: normalize, similarity (WMMA), loss ---
    k_rownorm<<<NROI, 256, 0, stream>>>(proj, invn);
    k_nfA<<<(NROI * PROJD + 255) / 256, 256, 0, stream>>>(proj, invn, nfA);
    k_nfB<<<(60 * 8 * 512 + 255) / 256, 256, 0, stream>>>(proj, invn, nfB);
    k_gemm<<<dim3(60, 60), 32, 0, stream>>>(nfA, nfB, nullptr, 256, 8, NROI,
                                            1.f / TEMPER, 0, nullptr, sim);
    k_lossrow<<<NROI, 256, 0, stream>>>(sim, labels, valid, lrow, arow);
    k_final<<<1, 256, 0, stream>>>(lrow, arow, out);
}
